// Criterion_66554813219062
// MI455X (gfx1250) — matrix operand order repaired and checked by reading, not compiled, not run on hardware
//
#include <hip/hip_runtime.h>
#include <hip/hip_bf16.h>

#define CAP      8000
#define T_FULL   32000
#define EPS      1e-10f
#define NTHREADS 256
#define CHUNK    32          // 250 active chunks cover CAP exactly

typedef __attribute__((ext_vector_type(16))) _Float16 v16h;
typedef __attribute__((ext_vector_type(8)))  float    v8f;

// ---------------------------------------------------------------------------
// 256-wide inclusive prefix sum executed by one full wave (threads 0..31)
// using 3x v_wmma_f32_16x16x32_f16. Values s[0..255] viewed as S[m][n]=s[16m+n].
//   T1 = S * U        (U[k][n]=1 if k<=n)        -> within-row inclusive prefix
//   T2 = S * J        (J[k][n]=1)                -> row sums broadcast
//   P  = Lstrict * T2 + T1 (L[m][k]=1 if k<m)    -> add carry of prior rows
// f16 A/B operands share contiguous k = i + 16*hi; f32 C/D layout per CDNA5 ISA 7.12.2.
// ---------------------------------------------------------------------------
__device__ inline void wave_scan256(const float* __restrict__ s,
                                    float* __restrict__ pre) {
  const int l  = (int)(threadIdx.x & 31);
  const int m  = l & 15;
  const bool hi = l >= 16;

  // A-layout (16x32 f16): lane l, half i -> M=l%16, K = i + 16*hi
  v16h a, la;
  v16h bu, bj;
#pragma unroll
  for (int i = 0; i < 16; ++i) {
    const int k = i + (hi ? 16 : 0);
    a[i]  = (_Float16)((k < 16) ? s[16 * m + k] : 0.0f);              // Swide
    la[i] = (_Float16)((k < 16 && k < m) ? 1.0f : 0.0f);              // Lstrict
    // B-layout (32x16 f16): lane l, half i -> N=l%16, K = i + 16*hi
    const int K = i + (hi ? 16 : 0);
    bu[i] = (_Float16)((K < 16 && K <= m) ? 1.0f : 0.0f);             // U
    bj[i] = (_Float16)((K < 16) ? 1.0f : 0.0f);                       // J
  }

  v8f zero = {};
  v8f t1 = __builtin_amdgcn_wmma_f32_16x16x32_f16(false, a, false, bu,
                                                  (short)0, zero, false, false);
  v8f t2 = __builtin_amdgcn_wmma_f32_16x16x32_f16(false, a, false, bj,
                                                  (short)0, zero, false, false);

  // Re-shape T2 (C layout: lane l, slot v -> M=v+8*hi, N=l%16) into B layout.
  // Lane n (<16) needs halfs 0..7 = T2[0..7][n] (own regs) and
  // halfs 8..15 = T2[8..15][n] (regs of lane n+16). Lanes >=16 feed K>=16,
  // which multiply zero columns of Lstrict -> zero them.
  v16h bt2;
#pragma unroll
  for (int j = 0; j < 8; ++j) {
    const float own   = t2[j];
    const float other = __shfl(t2[j], m + 16, 32);
    bt2[j]     = hi ? (_Float16)0.0f : (_Float16)own;
    bt2[j + 8] = hi ? (_Float16)0.0f : (_Float16)other;
  }

  v8f p = __builtin_amdgcn_wmma_f32_16x16x32_f16(false, la, false, bt2,
                                                 (short)0, t1, false, false);
#pragma unroll
  for (int v = 0; v < 8; ++v)
    pre[16 * (v + (hi ? 8 : 0)) + m] = p[v];
}

// ---------------------------------------------------------------------------
__device__ inline float block_reduce_sum(float v, float* scratch) {
  const int t = (int)threadIdx.x;
  scratch[t] = v;
  __syncthreads();
  for (int s = NTHREADS / 2; s > 0; s >>= 1) {
    if (t < s) scratch[t] += scratch[t + s];
    __syncthreads();
  }
  const float r = scratch[0];
  __syncthreads();
  return r;
}

__device__ inline float block_reduce_max(float v, float* scratch) {
  const int t = (int)threadIdx.x;
  scratch[t] = v;
  __syncthreads();
  for (int s = NTHREADS / 2; s > 0; s >>= 1) {
    if (t < s) scratch[t] = fmaxf(scratch[t], scratch[t + s]);
    __syncthreads();
  }
  const float r = scratch[0];
  __syncthreads();
  return r;
}

// ---------------------------------------------------------------------------
// One block per batch row. Single pass over HBM; squared samples for the
// first CAP positions are staged in LDS; suffix energy = E_total - prefix.
// ---------------------------------------------------------------------------
__global__ __launch_bounds__(NTHREADS)
void edc_row_kernel(const float* __restrict__ h, const float* __restrict__ g,
                    float* __restrict__ partial) {
  extern __shared__ float smem[];
  float* pw_h     = smem;                    // [CAP]
  float* pw_g     = smem + CAP;              // [CAP]
  float* chunkbuf = smem + 2 * CAP;          // [256] scan input / reduce scratch
  float* pre_h    = chunkbuf + NTHREADS;     // [256]
  float* pre_g    = pre_h + NTHREADS;        // [256]

  const int row = (int)blockIdx.x;
  const int t   = (int)threadIdx.x;
  const float* hr = h + (size_t)row * T_FULL;
  const float* gr = g + (size_t)row * T_FULL;

  // ---- Phase 1: streaming read, totals, last-nonzero, stash squares -------
  float sh = 0.f, sg = 0.f, lnh = -1.f, lng = -1.f;
  for (int i = t; i < T_FULL; i += NTHREADS) {
    const float x = hr[i], y = gr[i];
    const float px = x * x, py = y * y;
    sh += px; sg += py;
    if (x != 0.f) lnh = fmaxf(lnh, (float)i);
    if (y != 0.f) lng = fmaxf(lng, (float)i);
    if (i < CAP) { pw_h[i] = px; pw_g[i] = py; }
  }
  const float Eh = block_reduce_sum(sh, chunkbuf);
  const float Eg = block_reduce_sum(sg, chunkbuf);
  float Lh = block_reduce_max(lnh, chunkbuf);
  float Lg = block_reduce_max(lng, chunkbuf);
  const float inzh = (Lh >= 0.f) ? Lh : (float)(T_FULL - 1);
  const float inzg = (Lg >= 0.f) ? Lg : (float)(T_FULL - 1);

  // ---- Phase 2: chunk sums + WMMA cross-chunk scan -------------------------
  const int base = t * CHUNK;                // chunks 250..255 are empty
  float csh = 0.f, csg = 0.f;
  if (base < CAP) {
#pragma unroll
    for (int i = 0; i < CHUNK; ++i) { csh += pw_h[base + i]; csg += pw_g[base + i]; }
  }
  chunkbuf[t] = csh;
  __syncthreads();
  if (t < 32) wave_scan256(chunkbuf, pre_h);
  __syncthreads();
  chunkbuf[t] = csg;
  __syncthreads();
  if (t < 32) wave_scan256(chunkbuf, pre_g);
  __syncthreads();

  // ---- Phase 3: per-element dB, mask, |diff| accumulation ------------------
  float acc = 0.f;
  if (base < CAP) {
    const float offh = (t == 0) ? 0.f : pre_h[t - 1];   // exclusive carry-in
    const float offg = (t == 0) ? 0.f : pre_g[t - 1];
    const float dbh0 = log10f(Eh + EPS);
    const float dbg0 = log10f(Eg + EPS);
    float runh = 0.f, rung = 0.f;
#pragma unroll
    for (int i = 0; i < CHUNK; ++i) {
      const int idx = base + i;
      float eh = fmaxf(Eh - (offh + runh), 0.f);  runh += pw_h[idx];
      float eg = fmaxf(Eg - (offg + rung), 0.f);  rung += pw_g[idx];
      const float da = ((float)idx <= inzh) ? 10.f * (log10f(eh + EPS) - dbh0) : 0.f;
      const float db = ((float)idx <= inzg) ? 10.f * (log10f(eg + EPS) - dbg0) : 0.f;
      acc += fabsf(da - db);
    }
  }
  const float tot = block_reduce_sum(acc, chunkbuf);
  if (t == 0) partial[row] = tot;
}

__global__ void final_reduce(const float* __restrict__ partial,
                             float* __restrict__ out, int B) {
  __shared__ float s[NTHREADS];
  const int t = (int)threadIdx.x;
  float v = 0.f;
  for (int i = t; i < B; i += NTHREADS) v += partial[i];
  s[t] = v;
  __syncthreads();
  for (int k = NTHREADS / 2; k > 0; k >>= 1) {
    if (t < k) s[t] += s[t + k];
    __syncthreads();
  }
  if (t == 0) out[0] = s[0] / ((float)B * (float)CAP);
}

extern "C" void kernel_launch(void* const* d_in, const int* in_sizes, int n_in,
                              void* d_out, int out_size, void* d_ws, size_t ws_size,
                              hipStream_t stream) {
  const float* h = (const float*)d_in[0];
  const float* g = (const float*)d_in[1];
  float* out     = (float*)d_out;
  float* partial = (float*)d_ws;

  const int B = in_sizes[0] / T_FULL;
  const size_t smem_bytes = (size_t)(2 * CAP + 3 * NTHREADS) * sizeof(float);

  edc_row_kernel<<<B, NTHREADS, smem_bytes, stream>>>(h, g, partial);
  final_reduce<<<1, NTHREADS, 0, stream>>>(partial, out, B);
}
